// DEQDotProductAttentionTransformerMD17_16827681865897
// MI455X (gfx1250) — matrix operand order repaired and checked by reading, not compiled, not run on hardware
//
#include <hip/hip_runtime.h>
#include <math.h>

typedef __attribute__((ext_vector_type(16))) _Float16 v16h;
typedef __attribute__((ext_vector_type(8)))  float    v8f;
typedef _Float16 h16;

#define C0 128
#define C1 64
#define C2 32
#define N_NODES 4000
#define N_EDGES 64000
#define N_GRAPH 100
#define NWTP_ 416
#define NWDEG_ 224

#define ACT_NONE 0
#define ACT_SILU 1

__device__ __forceinline__ float fast_sigmoid(float x) {
  return __builtin_amdgcn_rcpf(1.0f + __expf(-x));
}

// ---------------------------------------------------------------------------
// utility kernels
// ---------------------------------------------------------------------------
__global__ void fill_kernel(float* p, float v, int n) {
  int i = blockIdx.x * blockDim.x + threadIdx.x;
  if (i < n) p[i] = v;
}

__global__ void cvt16_kernel(const float* __restrict__ s, h16* __restrict__ d, int n) {
  int i = blockIdx.x * blockDim.x + threadIdx.x;
  if (i < n) d[i] = (h16)s[i];
}

__global__ void add_kernel(float* __restrict__ y, const float* __restrict__ a,
                           const float* __restrict__ b, int n) {
  int i = blockIdx.x * blockDim.x + threadIdx.x;
  if (i < n) y[i] = a[i] + b[i];
}

// ---------------------------------------------------------------------------
// pack fp32 K x N weight matrix into WMMA B-fragment order (f16):
// P[((tn*KT + kt)*32 + lane)*16 + i] = W[(kt*32 + (lane<16?0:8) + (i<8?i:i+8))*N + tn*16 + (lane&15)]
// so each lane's v16h fragment is 32 contiguous bytes (2 x b128 loads).
// ---------------------------------------------------------------------------
__global__ void packB_kernel(const float* __restrict__ W, h16* __restrict__ P,
                             int K, int N) {
  int i = blockIdx.x * blockDim.x + threadIdx.x;
  if (i >= K * N) return;
  const int KT = K >> 5;
  const int ii = i & 15;
  const int lane = (i >> 4) & 31;
  const int blk = i >> 9;       // tn*KT + kt
  const int kt = blk % KT;
  const int tn = blk / KT;
  const int n = tn * 16 + (lane & 15);
  const int k = kt * 32 + ((lane < 16) ? 0 : 8) + ((ii < 8) ? ii : ii + 8);
  P[i] = (h16)W[(size_t)k * N + n];
}

// ---------------------------------------------------------------------------
// edge geometry: vec, r, sph harmonics, RBF (f16 for WMMA A operand)
// ---------------------------------------------------------------------------
__global__ __launch_bounds__(128) void edge_geom_kernel(
    const float* __restrict__ pos, const int* __restrict__ esrc,
    const int* __restrict__ edst, float* __restrict__ s1e,
    float* __restrict__ s2e, h16* __restrict__ rbfh) {
  const int e = blockIdx.x;
  const int t = threadIdx.x;
  const int s = esrc[e], d = edst[e];
  const float vx = pos[s * 3 + 0] - pos[d * 3 + 0];
  const float vy = pos[s * 3 + 1] - pos[d * 3 + 1];
  const float vz = pos[s * 3 + 2] - pos[d * 3 + 2];
  const float r = sqrtf(vx * vx + vy * vy + vz * vz + 1e-12f);
  const float ux = vx / r, uy = vy / r, uz = vz / r;
  if (t == 0) {
    const float s3 = 1.7320508075688772f;   // sqrt(3)
    const float s15 = 3.872983346207417f;   // sqrt(15)
    const float s5h = 1.118033988749895f;   // sqrt(5)/2
    const float s15h = 1.9364916731037085f; // sqrt(15)/2
    s1e[e * 3 + 0] = s3 * ux;
    s1e[e * 3 + 1] = s3 * uy;
    s1e[e * 3 + 2] = s3 * uz;
    s2e[e * 5 + 0] = s15 * ux * uy;
    s2e[e * 5 + 1] = s15 * uy * uz;
    s2e[e * 5 + 2] = s5h * (3.0f * uz * uz - 1.0f);
    s2e[e * 5 + 3] = s15 * ux * uz;
    s2e[e * 5 + 4] = s15h * (ux * ux - uy * uy);
  }
  const float mu = 5.0f * (float)t / 127.0f;
  const float inv_sig = 128.0f / 5.0f;
  const float z = (r - mu) * inv_sig;
  rbfh[(size_t)e * 128 + t] = (h16)__expf(-0.5f * z * z);
}

// ---------------------------------------------------------------------------
// WMMA GEMM: C[M,N] = act(A[M,K] @ W[K,N] + bias) + res ; optional f16 copy.
// No LDS: A-fragment = 2 contiguous b128 loads from row-major A (ISA layout:
// lane l holds row l&15, K in {kb..kb+7} u {kb+16..kb+23}, kb=(l<16?0:8));
// B-fragment = 2 contiguous b128 loads from the pre-packed weight buffer.
// Each wave computes a 16x32 output tile (2 N-tiles per A-fragment),
// 4 independent waves (M-tiles) per block. M,K mult 32-friendly; N mult 32.
// ---------------------------------------------------------------------------
union frag16 { v16h v; float4 q[2]; };

__global__ __launch_bounds__(128) void gemm_wmma_kernel(
    const h16* __restrict__ A, const h16* __restrict__ Wp,
    const float* __restrict__ bias, const float* __restrict__ res,
    float* __restrict__ C, h16* __restrict__ Ch, int M, int N, int K, int act) {
  const int lane = threadIdx.x & 31;
  const int wave = threadIdx.x >> 5;
  const int tileM = blockIdx.y * 4 + wave;
  if (tileM * 16 >= M) return;
  const int tn0 = blockIdx.x * 2;
  const int KT = K >> 5;

  const int row = tileM * 16 + (lane & 15);
  const int kb = (lane < 16) ? 0 : 8;
  const h16* Arow = A + (size_t)row * K + kb;
  const h16* B0 = Wp + ((size_t)tn0 * KT) * 512 + lane * 16;
  const h16* B1 = B0 + (size_t)KT * 512;

  v8f acc0 = {}, acc1 = {};
  for (int kt = 0; kt < KT; ++kt) {
    frag16 a, b0, b1;
    const h16* ap = Arow + kt * 32;
    a.q[0] = *reinterpret_cast<const float4*>(ap);
    a.q[1] = *reinterpret_cast<const float4*>(ap + 16);
    const h16* bp0 = B0 + (size_t)kt * 512;
    b0.q[0] = *reinterpret_cast<const float4*>(bp0);
    b0.q[1] = *reinterpret_cast<const float4*>(bp0 + 8);
    const h16* bp1 = B1 + (size_t)kt * 512;
    b1.q[0] = *reinterpret_cast<const float4*>(bp1);
    b1.q[1] = *reinterpret_cast<const float4*>(bp1 + 8);
    if (kt + 1 < KT) __builtin_prefetch(ap + 32, 0, 1); // global_prefetch
    acc0 = __builtin_amdgcn_wmma_f32_16x16x32_f16(false, a.v, false, b0.v,
                                                  (short)0, acc0, false, false);
    acc1 = __builtin_amdgcn_wmma_f32_16x16x32_f16(false, a.v, false, b1.v,
                                                  (short)0, acc1, false, false);
  }

  // C/D layout: lane<16 -> N=lane, M=v ; lane>=16 -> N=lane-16, M=8+v
  const int colA = tn0 * 16 + (lane & 15);
  const int colB = colA + 16;
  const int rb = tileM * 16 + ((lane >> 4) << 3);
  const float bvA = bias ? bias[colA] : 0.0f;
  const float bvB = bias ? bias[colB] : 0.0f;
#pragma unroll
  for (int v = 0; v < 8; ++v) {
    const size_t ia = (size_t)(rb + v) * N + colA;
    const size_t ib = (size_t)(rb + v) * N + colB;
    float xa = acc0[v] + bvA;
    float xb = acc1[v] + bvB;
    if (act == ACT_SILU) {
      xa = xa * fast_sigmoid(xa);
      xb = xb * fast_sigmoid(xb);
    }
    if (res) { xa += res[ia]; xb += res[ib]; }
    if (C) { C[ia] = xa; C[ib] = xb; }
    if (Ch) { Ch[ia] = (h16)xa; Ch[ib] = (h16)xb; }
  }
}

// ---------------------------------------------------------------------------
// degree embedding aggregation (segment_sum over dst with sph outer products)
// ---------------------------------------------------------------------------
__global__ void deg_agg_kernel(const float* __restrict__ wd,
                               const float* __restrict__ s1e,
                               const float* __restrict__ s2e,
                               const int* __restrict__ edst, float* a0,
                               float* a1, float* a2) {
  int i = blockIdx.x * blockDim.x + threadIdx.x;
  if (i >= N_EDGES * NWDEG_) return;
  const int e = i / NWDEG_, c = i % NWDEG_;
  const int d = edst[e];
  const float v = wd[i];
  if (c < C0) {
    atomicAdd(&a0[d * C0 + c], v);
  } else if (c < C0 + C1) {
    const int c1 = c - C0;
    for (int m = 0; m < 3; ++m)
      atomicAdd(&a1[(d * C1 + c1) * 3 + m], v * s1e[e * 3 + m]);
  } else {
    const int c2 = c - C0 - C1;
    for (int m = 0; m < 5; ++m)
      atomicAdd(&a2[(d * C2 + c2) * 5 + m], v * s2e[e * 5 + m]);
  }
}

__global__ void make_inj_kernel(const float* __restrict__ atom_embed,
                                const int* __restrict__ node_atom, float* inj0,
                                float* inj1, float* inj2, float inv) {
  int i = blockIdx.x * blockDim.x + threadIdx.x;
  if (i >= N_NODES * 480) return;
  const int n = i / 480, j = i % 480;
  if (j < C0) {
    const int idx = n * C0 + j;
    inj0[idx] = atom_embed[node_atom[n] * C0 + j] + inv * inj0[idx];
  } else if (j < C0 + C1 * 3) {
    const int idx = n * (C1 * 3) + (j - C0);
    inj1[idx] *= inv;
  } else {
    const int idx = n * (C2 * 5) + (j - C0 - C1 * 3);
    inj2[idx] *= inv;
  }
}

// ---------------------------------------------------------------------------
// equivariant layernorm (ln1/ln2): one node per 128-thread block
// ---------------------------------------------------------------------------
__global__ __launch_bounds__(128) void eqln_kernel(
    const float* __restrict__ h0, const float* __restrict__ h1,
    const float* __restrict__ h2, const float* __restrict__ b0,
    const float* __restrict__ g0, const float* __restrict__ g1,
    const float* __restrict__ g2, float* __restrict__ x0, h16* __restrict__ x0h,
    float* __restrict__ x1, float* __restrict__ x2) {
  const int n = blockIdx.x, t = threadIdx.x;
  __shared__ float sd[128];
  const float v = h0[n * C0 + t];
  sd[t] = v; __syncthreads();
  for (int s = 64; s > 0; s >>= 1) { if (t < s) sd[t] += sd[t + s]; __syncthreads(); }
  const float mu = sd[0] * (1.0f / C0);
  __syncthreads();
  const float dv = v - mu;
  sd[t] = dv * dv; __syncthreads();
  for (int s = 64; s > 0; s >>= 1) { if (t < s) sd[t] += sd[t + s]; __syncthreads(); }
  const float rstd = rsqrtf(sd[0] * (1.0f / C0) + 1e-5f);
  __syncthreads();
  const float y0 = dv * rstd * g0[t] + b0[t];
  x0[n * C0 + t] = y0;
  x0h[n * C0 + t] = (h16)y0;

  float acc = 0.0f;
  if (t < C1)
    for (int m = 0; m < 3; ++m) { float a = h1[(n * C1 + t) * 3 + m]; acc += a * a; }
  sd[t] = (t < C1) ? acc : 0.0f; __syncthreads();
  for (int s = 64; s > 0; s >>= 1) { if (t < s) sd[t] += sd[t + s]; __syncthreads(); }
  const float rs1 = rsqrtf(sd[0] * (1.0f / C1) + 1e-5f);
  __syncthreads();
  if (t < C1)
    for (int m = 0; m < 3; ++m)
      x1[(n * C1 + t) * 3 + m] = h1[(n * C1 + t) * 3 + m] * rs1 * g1[t];

  acc = 0.0f;
  if (t < C2)
    for (int m = 0; m < 5; ++m) { float a = h2[(n * C2 + t) * 5 + m]; acc += a * a; }
  sd[t] = (t < C2) ? acc : 0.0f; __syncthreads();
  for (int s = 64; s > 0; s >>= 1) { if (t < s) sd[t] += sd[t + s]; __syncthreads(); }
  const float rs2 = rsqrtf(sd[0] * (1.0f / C2) + 1e-5f);
  if (t < C2)
    for (int m = 0; m < 5; ++m)
      x2[(n * C2 + t) * 5 + m] = h2[(n * C2 + t) * 5 + m] * rs2 * g2[t];
}

// ---------------------------------------------------------------------------
// vector-channel linear (einsum ncm,cd->ndm) with optional sigmoid gate + res
// ---------------------------------------------------------------------------
__global__ void veclin_kernel(const float* __restrict__ X,
                              const float* __restrict__ W,
                              const float* __restrict__ gate, int gstride,
                              const float* __restrict__ res, float* __restrict__ Y,
                              int nN, int cIn, int dOut, int mD) {
  int i = blockIdx.x * blockDim.x + threadIdx.x;
  const int tot = nN * dOut * mD;
  if (i >= tot) return;
  const int m = i % mD;
  const int dd = (i / mD) % dOut;
  const int n = i / (mD * dOut);
  float acc = 0.0f;
  for (int c = 0; c < cIn; ++c) acc += X[(n * cIn + c) * mD + m] * W[c * dOut + dd];
  if (gate) acc *= fast_sigmoid(gate[n * gstride + dd]);
  if (res) acc += res[i];
  Y[i] = acc;
}

// ---------------------------------------------------------------------------
// attention: one wave per edge; lanes 8h..8h+7 own head h (120 dims/head)
// dtp computed in-registers; alpha = <q[dst], dtp(k[src])>/sqrt(120)
// ---------------------------------------------------------------------------
__global__ __launch_bounds__(256) void attn_alpha_kernel(
    const float* __restrict__ q0, const float* __restrict__ q1,
    const float* __restrict__ q2, const float* __restrict__ k0,
    const float* __restrict__ k1, const float* __restrict__ k2,
    const float* __restrict__ wtp, const float* __restrict__ s1e,
    const float* __restrict__ s2e, const int* __restrict__ esrc,
    const int* __restrict__ edst, float* __restrict__ alpha) {
  const int e = blockIdx.x * 8 + (threadIdx.x >> 5);
  if (e >= N_EDGES) return;
  const int lane = threadIdx.x & 31;
  const int hh = lane >> 3, t = lane & 7;
  const int s = esrc[e], d = edst[e];
  const float* w = wtp + (size_t)e * NWTP_;
  float S1[3], S2[5];
  for (int m = 0; m < 3; ++m) S1[m] = s1e[e * 3 + m];
  for (int m = 0; m < 5; ++m) S2[m] = s2e[e * 5 + m];
  float p = 0.0f;
#pragma unroll
  for (int i = 0; i < 4; ++i) {
    const int c = 32 * hh + t + 8 * i;
    const float kc = k0[s * C0 + c];
    float o0 = w[c] * kc;
    if (c < C1) {
      float dd = 0.0f;
      for (int m = 0; m < 3; ++m) dd += k1[(s * C1 + c) * 3 + m] * S1[m];
      o0 += w[C0 + c] * dd;
    }
    if (c < C2) {
      float dd = 0.0f;
      for (int m = 0; m < 5; ++m) dd += k2[(s * C2 + c) * 5 + m] * S2[m];
      o0 += w[192 + c] * dd;
    }
    p += q0[d * C0 + c] * o0;
  }
#pragma unroll
  for (int i = 0; i < 2; ++i) {
    const int c = 16 * hh + t + 8 * i;
    const float kc = k0[s * C0 + c];
    for (int m = 0; m < 3; ++m) {
      const float o1 = w[288 + c] * k1[(s * C1 + c) * 3 + m] + w[224 + c] * kc * S1[m];
      p += q1[(d * C1 + c) * 3 + m] * o1;
    }
  }
  {
    const int c = 8 * hh + t;
    const float kc = k0[s * C0 + c];
    for (int m = 0; m < 5; ++m) {
      const float o2 = w[384 + c] * k2[(s * C2 + c) * 5 + m] + w[352 + c] * kc * S2[m];
      p += q2[(d * C2 + c) * 5 + m] * o2;
    }
  }
  p += __shfl_xor(p, 1, 32);
  p += __shfl_xor(p, 2, 32);
  p += __shfl_xor(p, 4, 32);
  if (t == 0) alpha[e * 4 + hh] = p * 0.09128709291752768f; // 1/sqrt(120)
}

__device__ __forceinline__ void atomicMaxF(float* addr, float v) {
  if (v >= 0.0f) atomicMax((int*)addr, __float_as_int(v));
  else atomicMin((unsigned int*)addr, (unsigned int)__float_as_int(v));
}

__global__ void attn_max_kernel(const float* __restrict__ alpha,
                                const int* __restrict__ edst, float* mbuf) {
  int i = blockIdx.x * blockDim.x + threadIdx.x;
  if (i >= N_EDGES * 4) return;
  const int e = i >> 2, hh = i & 3;
  atomicMaxF(&mbuf[edst[e] * 4 + hh], alpha[i]);
}

__global__ void attn_exp_kernel(const float* __restrict__ alpha,
                                const int* __restrict__ edst,
                                const float* __restrict__ mbuf, float* ea,
                                float* ssum) {
  int i = blockIdx.x * blockDim.x + threadIdx.x;
  if (i >= N_EDGES * 4) return;
  const int e = i >> 2, hh = i & 3;
  const int d = edst[e];
  const float v = __expf(alpha[i] - mbuf[d * 4 + hh]);
  ea[i] = v;
  atomicAdd(&ssum[d * 4 + hh], v);
}

__global__ __launch_bounds__(256) void attn_msg_kernel(
    const float* __restrict__ v0, const float* __restrict__ v1,
    const float* __restrict__ v2, const float* __restrict__ wtp,
    const float* __restrict__ s1e, const float* __restrict__ s2e,
    const int* __restrict__ esrc, const int* __restrict__ edst,
    const float* __restrict__ ea, const float* __restrict__ ssum, float* o0,
    float* o1, float* o2) {
  const int e = blockIdx.x * 8 + (threadIdx.x >> 5);
  if (e >= N_EDGES) return;
  const int lane = threadIdx.x & 31;
  const int hh = lane >> 3, t = lane & 7;
  const int s = esrc[e], d = edst[e];
  const float* w = wtp + (size_t)e * NWTP_;
  const float a = ea[e * 4 + hh] * __builtin_amdgcn_rcpf(ssum[d * 4 + hh] + 1e-12f);
  float S1[3], S2[5];
  for (int m = 0; m < 3; ++m) S1[m] = s1e[e * 3 + m];
  for (int m = 0; m < 5; ++m) S2[m] = s2e[e * 5 + m];
#pragma unroll
  for (int i = 0; i < 4; ++i) {
    const int c = 32 * hh + t + 8 * i;
    const float vc = v0[s * C0 + c];
    float oo = w[c] * vc;
    if (c < C1) {
      float dd = 0.0f;
      for (int m = 0; m < 3; ++m) dd += v1[(s * C1 + c) * 3 + m] * S1[m];
      oo += w[C0 + c] * dd;
    }
    if (c < C2) {
      float dd = 0.0f;
      for (int m = 0; m < 5; ++m) dd += v2[(s * C2 + c) * 5 + m] * S2[m];
      oo += w[192 + c] * dd;
    }
    atomicAdd(&o0[d * C0 + c], a * oo);
  }
#pragma unroll
  for (int i = 0; i < 2; ++i) {
    const int c = 16 * hh + t + 8 * i;
    const float vc = v0[s * C0 + c];
    for (int m = 0; m < 3; ++m) {
      const float oo = w[288 + c] * v1[(s * C1 + c) * 3 + m] + w[224 + c] * vc * S1[m];
      atomicAdd(&o1[(d * C1 + c) * 3 + m], a * oo);
    }
  }
  {
    const int c = 8 * hh + t;
    const float vc = v0[s * C0 + c];
    for (int m = 0; m < 5; ++m) {
      const float oo = w[384 + c] * v2[(s * C2 + c) * 5 + m] + w[352 + c] * vc * S2[m];
      atomicAdd(&o2[(d * C2 + c) * 5 + m], a * oo);
    }
  }
}

// ---------------------------------------------------------------------------
// FFN gate: g0 = silu(sc[:, :C0]) as f16 (WMMA A operand)
// ---------------------------------------------------------------------------
__global__ void gates_kernel(const float* __restrict__ sc, h16* __restrict__ g0h) {
  int i = blockIdx.x * blockDim.x + threadIdx.x;
  if (i >= N_NODES * C0) return;
  const int n = i / C0, c = i % C0;
  const float v = sc[n * NWDEG_ + c];
  g0h[i] = (h16)(v * fast_sigmoid(v));
}

// ---------------------------------------------------------------------------
// LayerNorm over 512 features
// ---------------------------------------------------------------------------
__global__ __launch_bounds__(512) void ln512_kernel(const float* __restrict__ x,
                                                    const float* __restrict__ g,
                                                    const float* __restrict__ b,
                                                    h16* __restrict__ yh) {
  const int n = blockIdx.x, t = threadIdx.x;
  __shared__ float sd[512];
  const float v = x[(size_t)n * 512 + t];
  sd[t] = v; __syncthreads();
  for (int s = 256; s > 0; s >>= 1) { if (t < s) sd[t] += sd[t + s]; __syncthreads(); }
  const float mu = sd[0] * (1.0f / 512.0f);
  __syncthreads();
  const float dv = v - mu;
  sd[t] = dv * dv; __syncthreads();
  for (int s = 256; s > 0; s >>= 1) { if (t < s) sd[t] += sd[t + s]; __syncthreads(); }
  const float rstd = rsqrtf(sd[0] * (1.0f / 512.0f) + 1e-5f);
  yh[(size_t)n * 512 + t] = (h16)(dv * rstd * g[t] + b[t]);
}

// final scalar head + segment_sum over graphs
__global__ void head_kernel(const float* __restrict__ t2,
                            const float* __restrict__ w2,
                            const float* __restrict__ b2,
                            const int* __restrict__ batch, float* __restrict__ out,
                            float scale) {
  int n = blockIdx.x * blockDim.x + threadIdx.x;
  if (n >= N_NODES) return;
  float acc = b2[0];
  for (int i = 0; i < 512; ++i) acc += t2[(size_t)n * 512 + i] * w2[i];
  atomicAdd(&out[batch[n]], acc * scale);
}

// ---------------------------------------------------------------------------
// host side
// ---------------------------------------------------------------------------
// param-leaf indices assuming jax pytree (sorted dict keys) flattening,
// top-level insertion order: pos, params..., node_atom, batch, edge_src, edge_dst
enum {
  IN_POS = 0,
  IN_ATOM_EMBED = 1,
  IN_B_FCK_B1 = 2, IN_B_FCK_B2, IN_B_FCK_B3, IN_B_FCK_W1, IN_B_FCK_W2, IN_B_FCK_W3,
  IN_B_FCV_B1 = 8, IN_B_FCV_B2, IN_B_FCV_B3, IN_B_FCV_W1, IN_B_FCV_W2, IN_B_FCV_W3,
  IN_B_FFN_BS = 14,
  IN_B_FFNOUT_W0 = 15, IN_B_FFNOUT_W1, IN_B_FFNOUT_W2,
  IN_B_FFN_W1 = 18, IN_B_FFN_W2,
  IN_B_FFN_WS = 20,
  IN_B_K_W0 = 21, IN_B_K_W1, IN_B_K_W2,
  IN_B_LN1_B0 = 24, IN_B_LN1_G0, IN_B_LN1_G1, IN_B_LN1_G2,
  IN_B_LN2_B0 = 28, IN_B_LN2_G0, IN_B_LN2_G1, IN_B_LN2_G2,
  IN_B_PROJ_W0 = 32, IN_B_PROJ_W1, IN_B_PROJ_W2,
  IN_B_Q_W0 = 35, IN_B_Q_W1, IN_B_Q_W2,
  IN_B_V_W0 = 38, IN_B_V_W1, IN_B_V_W2,
  IN_FCD_B1 = 41, IN_FCD_B2, IN_FCD_B3, IN_FCD_W1, IN_FCD_W2, IN_FCD_W3,
  IN_F_B1 = 47, IN_F_B2,
  IN_F_FCK_B1 = 49, IN_F_FCK_B2, IN_F_FCK_B3, IN_F_FCK_W1, IN_F_FCK_W2, IN_F_FCK_W3,
  IN_F_FCV_B1 = 55, IN_F_FCV_B2, IN_F_FCV_B3, IN_F_FCV_W1, IN_F_FCV_W2, IN_F_FCV_W3,
  IN_F_K_W0 = 61, IN_F_K_W1, IN_F_K_W2,
  IN_F_LN1_B0 = 64, IN_F_LN1_G0, IN_F_LN1_G1, IN_F_LN1_G2,
  IN_F_LNB = 68, IN_F_LNG,
  IN_F_PROJS = 70,
  IN_F_Q_W0 = 71, IN_F_Q_W1, IN_F_Q_W2,
  IN_F_V_W0 = 74, IN_F_V_W1, IN_F_V_W2,
  IN_F_W1 = 77, IN_F_W2,
  IN_HEAD_B1 = 79, IN_HEAD_B2, IN_HEAD_W1, IN_HEAD_W2,
  IN_NORM_B = 83, IN_NORM_G,
  IN_NODE_ATOM = 85, IN_BATCH, IN_EDGE_SRC, IN_EDGE_DST
};

extern "C" void kernel_launch(void* const* d_in, const int* in_sizes, int n_in,
                              void* d_out, int out_size, void* d_ws, size_t ws_size,
                              hipStream_t stream) {
  (void)in_sizes; (void)n_in; (void)out_size; (void)ws_size;
  auto FPI = [&](int i) { return (const float*)d_in[i]; };
  auto IPI = [&](int i) { return (const int*)d_in[i]; };

  const float* pos = FPI(IN_POS);
  const int* node_atom = IPI(IN_NODE_ATOM);
  const int* batch = IPI(IN_BATCH);
  const int* esrc = IPI(IN_EDGE_SRC);
  const int* edst = IPI(IN_EDGE_DST);

  // -------- workspace layout --------
  char* cur = (char*)d_ws;
  auto alloc = [&](size_t bytes) -> void* {
    void* p = (void*)cur;
    cur += (bytes + 255) & ~(size_t)255;
    return p;
  };
  auto af = [&](size_t n) { return (float*)alloc(n * sizeof(float)); };
  auto ah = [&](size_t n) { return (h16*)alloc(n * sizeof(h16)); };

  // f16 packed weight copies (WMMA B-fragment order)
  h16 *W_fcd1 = ah(128 * 64), *W_fcd2 = ah(64 * 64), *W_fcd3 = ah(64 * 224);
  h16 *W_fck1 = ah(128 * 64), *W_fck2 = ah(64 * 64), *W_fck3 = ah(64 * 416);
  h16 *W_fcv1 = ah(128 * 64), *W_fcv2 = ah(64 * 64), *W_fcv3 = ah(64 * 416);
  h16 *W_Ffck1 = ah(128 * 64), *W_Ffck2 = ah(64 * 64), *W_Ffck3 = ah(64 * 416);
  h16 *W_Ffcv1 = ah(128 * 64), *W_Ffcv2 = ah(64 * 64), *W_Ffcv3 = ah(64 * 416);
  h16 *W_q0 = ah(128 * 128), *W_k0 = ah(128 * 128), *W_v0 = ah(128 * 128), *W_p0 = ah(128 * 128);
  h16 *W_Fq0 = ah(128 * 128), *W_Fk0 = ah(128 * 128), *W_Fv0 = ah(128 * 128);
  h16 *W_ffnws = ah(128 * 224), *W_ffnout0 = ah(128 * 128);
  h16 *W_projs = ah(128 * 512), *W_fw1 = ah(512 * 512), *W_fw2 = ah(512 * 512);
  h16* W_hw1 = ah(512 * 512);

  // edge buffers
  float* s1e = af((size_t)N_EDGES * 3);
  float* s2e = af((size_t)N_EDGES * 5);
  h16* rbfh = ah((size_t)N_EDGES * 128);
  h16* t64a = ah((size_t)N_EDGES * 64);
  h16* t64b = ah((size_t)N_EDGES * 64);
  float* wdE = af((size_t)N_EDGES * NWDEG_);
  float* wkE = af((size_t)N_EDGES * NWTP_);
  float* wvE = af((size_t)N_EDGES * NWTP_);
  float* alphaE = af((size_t)N_EDGES * 4);
  float* eaE = af((size_t)N_EDGES * 4);

  // node buffers
  float* inj0 = af(N_NODES * C0);
  float* inj1 = af(N_NODES * C1 * 3);
  float* inj2 = af(N_NODES * C2 * 5);
  float* z0 = af(N_NODES * C0);
  float* z1 = af(N_NODES * C1 * 3);
  float* z2 = af(N_NODES * C2 * 5);
  float* hb0 = af(N_NODES * C0);
  float* hb1 = af(N_NODES * C1 * 3);
  float* hb2 = af(N_NODES * C2 * 5);
  float* x0 = af(N_NODES * C0);
  h16* x0h = ah(N_NODES * C0);
  float* x1 = af(N_NODES * C1 * 3);
  float* x2 = af(N_NODES * C2 * 5);
  float* q0b = af(N_NODES * C0); float* k0b = af(N_NODES * C0); float* v0b = af(N_NODES * C0);
  float* q1b = af(N_NODES * C1 * 3); float* k1b = af(N_NODES * C1 * 3); float* v1b = af(N_NODES * C1 * 3);
  float* q2b = af(N_NODES * C2 * 5); float* k2b = af(N_NODES * C2 * 5); float* v2b = af(N_NODES * C2 * 5);
  float* mbuf = af(N_NODES * 4); float* ssum = af(N_NODES * 4);
  float* o0b = af(N_NODES * C0);
  float* o1b = af(N_NODES * C1 * 3);
  float* o2b = af(N_NODES * C2 * 5);
  h16* o0h = ah(N_NODES * C0);
  float* sc = af(N_NODES * NWDEG_);
  h16* g0h = ah(N_NODES * C0);
  float* v1g = af(N_NODES * C1 * 3);
  float* v2g = af(N_NODES * C2 * 5);
  float* s512 = af((size_t)N_NODES * 512);
  float* u512 = af((size_t)N_NODES * 512);
  float* t2f = af((size_t)N_NODES * 512);
  h16* zh = ah((size_t)N_NODES * 512);
  h16* th = ah((size_t)N_NODES * 512);
  h16* yh = ah((size_t)N_NODES * 512);

  // -------- launch helpers --------
  auto packB = [&](const float* s, h16* dst, int K, int N) {
    packB_kernel<<<dim3((K * N + 255) / 256), 256, 0, stream>>>(s, dst, K, N);
  };
  auto cvtbuf = [&](const float* s, h16* dst, int n) {
    cvt16_kernel<<<dim3((n + 255) / 256), 256, 0, stream>>>(s, dst, n);
  };
  auto fill = [&](float* p, float v, int n) {
    fill_kernel<<<dim3((n + 255) / 256), 256, 0, stream>>>(p, v, n);
  };
  auto addk = [&](float* y, const float* a, const float* b, int n) {
    add_kernel<<<dim3((n + 255) / 256), 256, 0, stream>>>(y, a, b, n);
  };
  auto gemm = [&](const h16* A, const h16* Wp, const float* bias, const float* res,
                  float* Cc, h16* Ch, int M, int N, int K, int act) {
    const int mTiles = M / 16;
    gemm_wmma_kernel<<<dim3(N / 32, (mTiles + 3) / 4), 128, 0, stream>>>(
        A, Wp, bias, res, Cc, Ch, M, N, K, act);
  };
  auto veclin = [&](const float* X, const float* W, const float* gate, int gstride,
                    const float* res, float* Y, int cIn, int dOut, int mD) {
    const int tot = N_NODES * dOut * mD;
    veclin_kernel<<<dim3((tot + 255) / 256), 256, 0, stream>>>(
        X, W, gate, gstride, res, Y, N_NODES, cIn, dOut, mD);
  };

  // -------- stage 0: weight conversion + packing (fp32 -> f16 fragments) ----
  packB(FPI(IN_FCD_W1), W_fcd1, 128, 64); packB(FPI(IN_FCD_W2), W_fcd2, 64, 64);
  packB(FPI(IN_FCD_W3), W_fcd3, 64, 224);
  packB(FPI(IN_B_FCK_W1), W_fck1, 128, 64); packB(FPI(IN_B_FCK_W2), W_fck2, 64, 64);
  packB(FPI(IN_B_FCK_W3), W_fck3, 64, 416);
  packB(FPI(IN_B_FCV_W1), W_fcv1, 128, 64); packB(FPI(IN_B_FCV_W2), W_fcv2, 64, 64);
  packB(FPI(IN_B_FCV_W3), W_fcv3, 64, 416);
  packB(FPI(IN_F_FCK_W1), W_Ffck1, 128, 64); packB(FPI(IN_F_FCK_W2), W_Ffck2, 64, 64);
  packB(FPI(IN_F_FCK_W3), W_Ffck3, 64, 416);
  packB(FPI(IN_F_FCV_W1), W_Ffcv1, 128, 64); packB(FPI(IN_F_FCV_W2), W_Ffcv2, 64, 64);
  packB(FPI(IN_F_FCV_W3), W_Ffcv3, 64, 416);
  packB(FPI(IN_B_Q_W0), W_q0, 128, 128); packB(FPI(IN_B_K_W0), W_k0, 128, 128);
  packB(FPI(IN_B_V_W0), W_v0, 128, 128); packB(FPI(IN_B_PROJ_W0), W_p0, 128, 128);
  packB(FPI(IN_F_Q_W0), W_Fq0, 128, 128); packB(FPI(IN_F_K_W0), W_Fk0, 128, 128);
  packB(FPI(IN_F_V_W0), W_Fv0, 128, 128);
  packB(FPI(IN_B_FFN_WS), W_ffnws, 128, 224);
  packB(FPI(IN_B_FFNOUT_W0), W_ffnout0, 128, 128);
  packB(FPI(IN_F_PROJS), W_projs, 128, 512);
  packB(FPI(IN_F_W1), W_fw1, 512, 512); packB(FPI(IN_F_W2), W_fw2, 512, 512);
  packB(FPI(IN_HEAD_W1), W_hw1, 512, 512);

  // -------- stage 1: edge geometry + RBF --------
  edge_geom_kernel<<<N_EDGES, 128, 0, stream>>>(pos, esrc, edst, s1e, s2e, rbfh);

  // -------- stage 2: edge MLPs (hoisted out of DEQ loop) --------
  gemm(rbfh, W_fcd1, FPI(IN_FCD_B1), nullptr, nullptr, t64a, N_EDGES, 64, 128, ACT_SILU);
  gemm(t64a, W_fcd2, FPI(IN_FCD_B2), nullptr, nullptr, t64b, N_EDGES, 64, 64, ACT_SILU);
  gemm(t64b, W_fcd3, FPI(IN_FCD_B3), nullptr, wdE, nullptr, N_EDGES, NWDEG_, 64, ACT_NONE);
  gemm(rbfh, W_fck1, FPI(IN_B_FCK_B1), nullptr, nullptr, t64a, N_EDGES, 64, 128, ACT_SILU);
  gemm(t64a, W_fck2, FPI(IN_B_FCK_B2), nullptr, nullptr, t64b, N_EDGES, 64, 64, ACT_SILU);
  gemm(t64b, W_fck3, FPI(IN_B_FCK_B3), nullptr, wkE, nullptr, N_EDGES, NWTP_, 64, ACT_NONE);
  gemm(rbfh, W_fcv1, FPI(IN_B_FCV_B1), nullptr, nullptr, t64a, N_EDGES, 64, 128, ACT_SILU);
  gemm(t64a, W_fcv2, FPI(IN_B_FCV_B2), nullptr, nullptr, t64b, N_EDGES, 64, 64, ACT_SILU);
  gemm(t64b, W_fcv3, FPI(IN_B_FCV_B3), nullptr, wvE, nullptr, N_EDGES, NWTP_, 64, ACT_NONE);

  // -------- stage 3: injection (degree embedding) --------
  fill(inj0, 0.0f, N_NODES * C0);
  fill(inj1, 0.0f, N_NODES * C1 * 3);
  fill(inj2, 0.0f, N_NODES * C2 * 5);
  deg_agg_kernel<<<dim3((N_EDGES * NWDEG_ + 255) / 256), 256, 0, stream>>>(
      wdE, s1e, s2e, edst, inj0, inj1, inj2);
  make_inj_kernel<<<dim3((N_NODES * 480 + 255) / 256), 256, 0, stream>>>(
      FPI(IN_ATOM_EMBED), node_atom, inj0, inj1, inj2, 1.0f / sqrtf(15.57f));

  // -------- attention helper --------
  auto attention = [&](float* ph0, float* ph1, float* ph2, const float* ln_b0,
                       const float* ln_g0, const float* ln_g1, const float* ln_g2,
                       const h16* wq0, const float* wq1, const float* wq2,
                       const h16* wk0, const float* wk1, const float* wk2,
                       const h16* wv0, const float* wv1, const float* wv2) {
    eqln_kernel<<<N_NODES, 128, 0, stream>>>(ph0, ph1, ph2, ln_b0, ln_g0, ln_g1,
                                             ln_g2, x0, x0h, x1, x2);
    gemm(x0h, wq0, nullptr, nullptr, q0b, nullptr, N_NODES, C0, C0, ACT_NONE);
    gemm(x0h, wk0, nullptr, nullptr, k0b, nullptr, N_NODES, C0, C0, ACT_NONE);
    gemm(x0h, wv0, nullptr, nullptr, v0b, nullptr, N_NODES, C0, C0, ACT_NONE);
    veclin(x1, wq1, nullptr, 0, nullptr, q1b, C1, C1, 3);
    veclin(x2, wq2, nullptr, 0, nullptr, q2b, C2, C2, 5);
    veclin(x1, wk1, nullptr, 0, nullptr, k1b, C1, C1, 3);
    veclin(x2, wk2, nullptr, 0, nullptr, k2b, C2, C2, 5);
    veclin(x1, wv1, nullptr, 0, nullptr, v1b, C1, C1, 3);
    veclin(x2, wv2, nullptr, 0, nullptr, v2b, C2, C2, 5);
    attn_alpha_kernel<<<N_EDGES / 8, 256, 0, stream>>>(q0b, q1b, q2b, k0b, k1b,
                                                       k2b, wkE, s1e, s2e, esrc,
                                                       edst, alphaE);
    fill(mbuf, -1e30f, N_NODES * 4);
    fill(ssum, 0.0f, N_NODES * 4);
    attn_max_kernel<<<dim3((N_EDGES * 4 + 255) / 256), 256, 0, stream>>>(alphaE, edst, mbuf);
    attn_exp_kernel<<<dim3((N_EDGES * 4 + 255) / 256), 256, 0, stream>>>(alphaE, edst, mbuf, eaE, ssum);
    fill(o0b, 0.0f, N_NODES * C0);
    fill(o1b, 0.0f, N_NODES * C1 * 3);
    fill(o2b, 0.0f, N_NODES * C2 * 5);
    attn_msg_kernel<<<N_EDGES / 8, 256, 0, stream>>>(v0b, v1b, v2b, wvE, s1e, s2e,
                                                     esrc, edst, eaE, ssum, o0b,
                                                     o1b, o2b);
  };

  // -------- stage 4: DEQ loop --------
  float *pz0 = z0, *pz1 = z1, *pz2 = z2;
  float *ph0 = hb0, *ph1 = hb1, *ph2 = hb2;
  fill(pz0, 0.0f, N_NODES * C0);
  fill(pz1, 0.0f, N_NODES * C1 * 3);
  fill(pz2, 0.0f, N_NODES * C2 * 5);
  for (int it = 0; it < 4; ++it) {
    addk(ph0, pz0, inj0, N_NODES * C0);
    addk(ph1, pz1, inj1, N_NODES * C1 * 3);
    addk(ph2, pz2, inj2, N_NODES * C2 * 5);
    attention(ph0, ph1, ph2, FPI(IN_B_LN1_B0), FPI(IN_B_LN1_G0), FPI(IN_B_LN1_G1),
              FPI(IN_B_LN1_G2), W_q0, FPI(IN_B_Q_W1), FPI(IN_B_Q_W2), W_k0,
              FPI(IN_B_K_W1), FPI(IN_B_K_W2), W_v0, FPI(IN_B_V_W1), FPI(IN_B_V_W2));
    // proj + residual
    cvtbuf(o0b, o0h, N_NODES * C0);
    gemm(o0h, W_p0, nullptr, ph0, ph0, nullptr, N_NODES, C0, C0, ACT_NONE);
    veclin(o1b, FPI(IN_B_PROJ_W1), nullptr, 0, ph1, ph1, C1, C1, 3);
    veclin(o2b, FPI(IN_B_PROJ_W2), nullptr, 0, ph2, ph2, C2, C2, 5);
    // FFN
    eqln_kernel<<<N_NODES, 128, 0, stream>>>(ph0, ph1, ph2, FPI(IN_B_LN2_B0),
                                             FPI(IN_B_LN2_G0), FPI(IN_B_LN2_G1),
                                             FPI(IN_B_LN2_G2), x0, x0h, x1, x2);
    gemm(x0h, W_ffnws, FPI(IN_B_FFN_BS), nullptr, sc, nullptr, N_NODES, NWDEG_, C0, ACT_NONE);
    gates_kernel<<<dim3((N_NODES * C0 + 255) / 256), 256, 0, stream>>>(sc, g0h);
    veclin(x1, FPI(IN_B_FFN_W1), sc + C0, NWDEG_, nullptr, v1g, C1, C1, 3);
    veclin(x2, FPI(IN_B_FFN_W2), sc + C0 + C1, NWDEG_, nullptr, v2g, C2, C2, 5);
    gemm(g0h, W_ffnout0, nullptr, ph0, ph0, nullptr, N_NODES, C0, C0, ACT_NONE);
    veclin(v1g, FPI(IN_B_FFNOUT_W1), nullptr, 0, ph1, ph1, C1, C1, 3);
    veclin(v2g, FPI(IN_B_FFNOUT_W2), nullptr, 0, ph2, ph2, C2, C2, 5);
    float* tmp;
    tmp = pz0; pz0 = ph0; ph0 = tmp;
    tmp = pz1; pz1 = ph1; ph1 = tmp;
    tmp = pz2; pz2 = ph2; ph2 = tmp;
  }

  // -------- stage 5: final block --------
  gemm(rbfh, W_Ffck1, FPI(IN_F_FCK_B1), nullptr, nullptr, t64a, N_EDGES, 64, 128, ACT_SILU);
  gemm(t64a, W_Ffck2, FPI(IN_F_FCK_B2), nullptr, nullptr, t64b, N_EDGES, 64, 64, ACT_SILU);
  gemm(t64b, W_Ffck3, FPI(IN_F_FCK_B3), nullptr, wkE, nullptr, N_EDGES, NWTP_, 64, ACT_NONE);
  gemm(rbfh, W_Ffcv1, FPI(IN_F_FCV_B1), nullptr, nullptr, t64a, N_EDGES, 64, 128, ACT_SILU);
  gemm(t64a, W_Ffcv2, FPI(IN_F_FCV_B2), nullptr, nullptr, t64b, N_EDGES, 64, 64, ACT_SILU);
  gemm(t64b, W_Ffcv3, FPI(IN_F_FCV_B3), nullptr, wvE, nullptr, N_EDGES, NWTP_, 64, ACT_NONE);

  attention(pz0, pz1, pz2, FPI(IN_F_LN1_B0), FPI(IN_F_LN1_G0), FPI(IN_F_LN1_G1),
            FPI(IN_F_LN1_G2), W_Fq0, FPI(IN_F_Q_W1), FPI(IN_F_Q_W2), W_Fk0,
            FPI(IN_F_K_W1), FPI(IN_F_K_W2), W_Fv0, FPI(IN_F_V_W1), FPI(IN_F_V_W2));

  cvtbuf(o0b, o0h, N_NODES * C0);
  gemm(o0h, W_projs, nullptr, nullptr, s512, nullptr, N_NODES, 512, 128, ACT_NONE);
  ln512_kernel<<<N_NODES, 512, 0, stream>>>(s512, FPI(IN_F_LNG), FPI(IN_F_LNB), zh);
  gemm(zh, W_fw1, FPI(IN_F_B1), nullptr, nullptr, th, N_NODES, 512, 512, ACT_SILU);
  gemm(th, W_fw2, FPI(IN_F_B2), s512, u512, nullptr, N_NODES, 512, 512, ACT_NONE);
  ln512_kernel<<<N_NODES, 512, 0, stream>>>(u512, FPI(IN_NORM_G), FPI(IN_NORM_B), yh);
  gemm(yh, W_hw1, FPI(IN_HEAD_B1), nullptr, t2f, nullptr, N_NODES, 512, 512, ACT_SILU);

  fill((float*)d_out, 0.0f, N_GRAPH);
  head_kernel<<<dim3((N_NODES + 127) / 128), 128, 0, stream>>>(
      t2f, FPI(IN_HEAD_W2), FPI(IN_HEAD_B2), batch, (float*)d_out,
      1.0f / sqrtf(18.03f));
}
